// WeightBasedTripleLoss_9345848836159
// MI455X (gfx1250) — matrix operand order repaired and checked by reading, not compiled, not run on hardware
//
#include <hip/hip_runtime.h>

// Problem constants (from reference): x is [4096,128] f32, K=8 -> 7 negatives/anchor.
#define NN      4096
#define DD      128
#define KBATCH  8
#define NEG     (KBATCH - 1)          // 7
#define NTRIP   (NN * NEG)            // 28672
#define CUTOFF  1.4f
#define PD_EPS  1e-6f

typedef __bf16 bf16_t;
typedef __bf16 v16bf __attribute__((ext_vector_type(16)));
typedef float  v8f   __attribute__((ext_vector_type(8)));
typedef unsigned int v4u __attribute__((ext_vector_type(4)));
typedef int          v8i __attribute__((ext_vector_type(8)));
typedef int          v4i __attribute__((ext_vector_type(4)));

#define AS_GLOBAL __attribute__((address_space(1)))
#define AS_LDS    __attribute__((address_space(3)))

// ---- ordered-uint encoding for float atomic max (NaN encodes above +inf, matching jnp.max NaN propagation) ----
__device__ __forceinline__ unsigned encf(float f) {
    unsigned b = __float_as_uint(f);
    return (b & 0x80000000u) ? ~b : (b | 0x80000000u);
}
__device__ __forceinline__ float decf(unsigned e) {
    unsigned b = (e & 0x80000000u) ? (e & 0x7FFFFFFFu) : ~e;
    return __uint_as_float(b);
}

// ---- deterministic counter-based Gumbel noise (JAX threefry is not bit-reproducible here; keep it deterministic) ----
__device__ __forceinline__ float gumbel32(unsigned i, unsigned k, unsigned j) {
    unsigned h = i * 0x9E3779B1u ^ (k + 1u) * 0x85EBCA77u ^ j * 0xC2B2AE3Du ^ 0x2545F491u;
    h ^= h >> 16; h *= 0x7FEB352Du;
    h ^= h >> 15; h *= 0x846CA68Bu;
    h ^= h >> 16;
    float u = (float)(h >> 8) * (1.0f / 16777216.0f) + (0.5f / 16777216.0f); // (0,1)
    return -logf(-logf(u));
}

// ---------------------------------------------------------------------------
// K0: init accumulators (d_out/d_ws are poisoned by the harness; must re-init every call)
// ---------------------------------------------------------------------------
__global__ void k_init(unsigned* __restrict__ gmax_enc, float* __restrict__ out) {
    if (threadIdx.x == 0 && blockIdx.x == 0) {
        *gmax_enc = encf(-__builtin_inff()); // 0x007FFFFF
        out[0]    = 0.0f;
    }
}

// ---------------------------------------------------------------------------
// K1: row L2-normalize, emit bf16 xn for the WMMA GEMM + per-row sum(xn^2)
// ---------------------------------------------------------------------------
__global__ void k_normalize(const float* __restrict__ x,
                            bf16_t* __restrict__ xnb,
                            float* __restrict__ sq) {
    const int row = blockIdx.x;
    const int t   = threadIdx.x; // 128 threads, one per feature
    __shared__ float s[DD];

    float v = x[(size_t)row * DD + t];
    s[t] = v * v;
    __syncthreads();
    for (int o = DD / 2; o > 0; o >>= 1) {
        if (t < o) s[t] += s[t + o];
        __syncthreads();
    }
    const float norm = sqrtf(s[0]) + 1e-5f;
    __syncthreads();

    const float xv = v / norm;
    xnb[(size_t)row * DD + t] = (bf16_t)xv;

    s[t] = xv * xv;
    __syncthreads();
    for (int o = DD / 2; o > 0; o >>= 1) {
        if (t < o) s[t] += s[t + o];
        __syncthreads();
    }
    if (t == 0) sq[row] = s[0];
}

// ---------------------------------------------------------------------------
// K2: G = xn @ xn^T via v_wmma_f32_16x16x32_bf16.
// The shared 16x128 bf16 j-tile is DMA'd into LDS ONCE per block by the
// Tensor Data Mover (tensor_load_to_lds, issued by wave 0, TENSORcnt wait),
// then all 8 waves build B fragments from LDS. Fused epilogue:
// dist -> log_w -> global max (ordered-uint atomic) + masked log_w store.
// Uniform control flow at the WMMA => EXEC all-ones (ISA requirement).
// ---------------------------------------------------------------------------
__global__ void k_gram(const bf16_t* __restrict__ xnb,
                       const float* __restrict__ sq,
                       float* __restrict__ LW,
                       unsigned* __restrict__ gmax_enc) {
    const int lane  = threadIdx.x & 31;
    const int wave  = threadIdx.x >> 5;
    const int jBase = blockIdx.x * 16;
    const int iBase = (blockIdx.y * 8 + wave) * 16;
    const int m     = lane & 15;  // A: row M; C/D: col N
    const int grp   = lane >> 4;  // A: K-group; C/D: M-half select

    __shared__ __align__(16) bf16_t btile[16 * DD]; // 4 KB j-tile, shared by 8 waves

    // --- TDM: DMA rows [jBase, jBase+16) of xn (bf16) into LDS -----------------
    if (wave == 0) {
        const unsigned long long gaddr =
            (unsigned long long)(const void*)(xnb + (size_t)jBase * DD);
        const unsigned ldsaddr = (unsigned)(unsigned long long)(AS_LDS bf16_t*)btile;

        v4u g0;
        g0.x = 1u;                                   // count=1, user D#
        g0.y = ldsaddr;                              // lds_addr (bytes)
        g0.z = (unsigned)(gaddr & 0xFFFFFFFFull);    // global_addr[31:0]
        g0.w = (unsigned)((gaddr >> 32) & 0x01FFFFFFull) | (2u << 30); // addr[56:32] | type=2

        v8i g1;
        g1[0] = 0x00010000;            // data_size=1 (2 bytes), no multicast/pad/iterate
        g1[1] = (int)(128u << 16);     // tensor_dim0 = 128 (bits 79:48, low half)
        g1[2] = (int)(4096u << 16);    // tensor_dim1 = 4096 (bits 111:80, low half)
        g1[3] = (int)(128u << 16);     // tile_dim0 = 128 (bits 127:112)
        g1[4] = 16;                    // tile_dim1 = 16 rows
        g1[5] = 128;                   // tensor_dim0_stride = 128 elements
        g1[6] = 0;                     // dim0_stride hi | dim1_stride lo (unused, 2D)
        g1[7] = 0;
        const v4i zz4 = {0, 0, 0, 0};  // groups 2/3 unused (<=2D tensor)
        const v8i zz8 = {0, 0, 0, 0, 0, 0, 0, 0};

        __builtin_amdgcn_tensor_load_to_lds(g0, g1, zz4, zz4, zz8, 0);
        __builtin_amdgcn_s_wait_tensorcnt(0);
    }
    __syncthreads();

    const bf16_t* arow = xnb + (size_t)(iBase + m) * DD;
    __builtin_prefetch(arow, 0, 1); // -> global_prefetch_b8 (gfx1250)

    v8f C = {0.f, 0.f, 0.f, 0.f, 0.f, 0.f, 0.f, 0.f};
    #pragma unroll
    for (int kb = 0; kb < DD; kb += 32) {
        v16bf A, B;
        // A fragment: 16-bit 16x32 layout — lane(m,grp); VGPR v holds K pairs.
        #pragma unroll
        for (int v = 0; v < 8; ++v) {
            const int k = kb + ((v >> 2) << 4) + grp * 8 + ((v & 3) << 1);
            A[2 * v]     = arow[k];
            A[2 * v + 1] = arow[k + 1];
        }
        // B fragment from LDS tile: lane = N (m), element e = K (same order as A); B[k][n] = xn[jBase+n][k].
        #pragma unroll
        for (int e = 0; e < 16; ++e)
            B[e] = btile[m * DD + kb + ((e >> 3) << 4) + grp * 8 + (e & 7)];

        C = __builtin_amdgcn_wmma_f32_16x16x32_bf16(false, A, false, B,
                                                    (short)0, C, false, false);
    }

    // Epilogue: C/D layout — lane: N = lane&15, VGPR r: M = r + 8*(lane>>4)
    unsigned localEnc = 0u;
    const int   gj  = jBase + m;
    const float sqj = sq[gj];
    #pragma unroll
    for (int r = 0; r < 8; ++r) {
        const int gi = iBase + r + grp * 8;
        const float g    = C[r];
        const float d2   = sq[gi] + sqj - 2.0f * g + (gi == gj ? 1.0f : 0.0f);
        const float dist = sqrtf(d2);
        // log_w = (2-d) log(dist) - (d-3)/2 log(1 - dist^2/4), d = 128
        const float lw = -126.0f * logf(dist) - 62.5f * logf(1.0f - 0.25f * dist * dist);

        const unsigned e = encf(lw);
        localEnc = localEnc > e ? localEnc : e;

        const bool ok = ((gi >> 3) != (gj >> 3)) && (dist < CUTOFF);
        // keep NaN (mask multiply by 0 keeps NaN in the reference)
        const float val = (lw != lw) ? lw : (ok ? lw : -__builtin_inff());
        LW[(size_t)gi * NN + gj] = val;
    }

    __shared__ unsigned smax[256];
    smax[threadIdx.x] = localEnc;
    __syncthreads();
    for (int o = 128; o > 0; o >>= 1) {
        if (threadIdx.x < o) {
            const unsigned a = smax[threadIdx.x], b = smax[threadIdx.x + o];
            smax[threadIdx.x] = a > b ? a : b;
        }
        __syncthreads();
    }
    if (threadIdx.x == 0) atomicMax(gmax_enc, smax[0]);
}

// ---------------------------------------------------------------------------
// K3: per anchor row — stage log_w row into LDS via global_load_async_to_lds_b128
//     (ASYNCcnt path), row-sum/validity, 7 Gumbel-argmax negative draws,
//     triplet distances, partial-mean atomicAdd into d_out.
// ---------------------------------------------------------------------------
__global__ void k_sample_loss(const float* __restrict__ x,
                              const float* __restrict__ LW,
                              const unsigned* __restrict__ gmax_enc,
                              float* __restrict__ out) {
    const int i = blockIdx.x;
    const int t = threadIdx.x; // 256

    __shared__ __align__(16) float lrow[NN]; // 16 KB of the 320 KB WGP LDS
    __shared__ float redf[256];
    __shared__ int   redi[256];
    __shared__ int   s_nidx[NEG];

    const float gmax = decf(*gmax_enc);
    const float* Lr  = LW + (size_t)i * NN;

    // Async DMA: 1024 x b128 transfers, 4 per thread; wait ASYNCcnt then barrier.
    #pragma unroll
    for (int q = 0; q < NN / 4 / 256; ++q) {
        const int j4 = (t + q * 256) * 4;
        __builtin_amdgcn_global_load_async_to_lds_b128(
            (AS_GLOBAL v4i*)(Lr + j4),
            (AS_LDS v4i*)&lrow[j4], 0, 0);
    }
    __builtin_amdgcn_s_wait_asynccnt(0);
    __syncthreads();

    // row sum of w = exp(l - gmax); NaN/empty rows -> invalid -> uniform fallback
    float s = 0.0f;
    for (int j = t; j < NN; j += 256) s += expf(lrow[j] - gmax);
    redf[t] = s;
    __syncthreads();
    for (int o = 128; o > 0; o >>= 1) {
        if (t < o) redf[t] += redf[t + o];
        __syncthreads();
    }
    const bool valid = redf[0] > 0.0f;
    __syncthreads();

    // 7 Gumbel-max draws
    for (int k = 0; k < NEG; ++k) {
        float best = -__builtin_inff();
        int   bi   = NN;
        for (int j = t; j < NN; j += 256) {
            const float wv    = expf(lrow[j] - gmax);
            const float logit = (wv > 0.0f) ? logf(wv) : -__builtin_inff();
            const float key   = (valid ? logit : 0.0f) + gumbel32((unsigned)i, (unsigned)k, (unsigned)j);
            if (key > best || (key == best && j < bi)) { best = key; bi = j; }
        }
        redf[t] = best; redi[t] = bi;
        __syncthreads();
        for (int o = 128; o > 0; o >>= 1) {
            if (t < o) {
                const float a = redf[t], b = redf[t + o];
                if (b > a || (b == a && redi[t + o] < redi[t])) { redf[t] = b; redi[t] = redi[t + o]; }
            }
            __syncthreads();
        }
        if (t == 0) s_nidx[k] = (redi[0] < NN) ? redi[0] : 0;
        __syncthreads();
    }

    // triplet distances on the ORIGINAL x (pairwise eps inside the square, as in torch)
    float lsum  = 0.0f;
    const int r = i & (KBATCH - 1);
    for (int k = 0; k < NEG; ++k) {
        const int nidx = s_nidx[k];
        const int poff = (k < r) ? k : k + 1;
        const int pidx = (i & ~(KBATCH - 1)) + poff;

        float dap = 0.0f, dan = 0.0f;
        if (t < DD) {
            const float a  = x[(size_t)i * DD + t];
            const float dp = a - x[(size_t)pidx * DD + t] + PD_EPS;
            const float dn = a - x[(size_t)nidx * DD + t] + PD_EPS;
            dap = dp * dp; dan = dn * dn;
        }
        redf[t] = dap;
        __syncthreads();
        for (int o = 128; o > 0; o >>= 1) { if (t < o) redf[t] += redf[t + o]; __syncthreads(); }
        const float dapS = redf[0];
        __syncthreads();
        redf[t] = dan;
        __syncthreads();
        for (int o = 128; o > 0; o >>= 1) { if (t < o) redf[t] += redf[t + o]; __syncthreads(); }
        if (t == 0) lsum += fmaxf(sqrtf(dapS) - sqrtf(redf[0]) + 1.0f, 0.0f);
        __syncthreads();
    }
    if (t == 0) atomicAdd(out, lsum * (1.0f / (float)NTRIP));
}

// ---------------------------------------------------------------------------
extern "C" void kernel_launch(void* const* d_in, const int* in_sizes, int n_in,
                              void* d_out, int out_size, void* d_ws, size_t ws_size,
                              hipStream_t stream) {
    (void)in_sizes; (void)n_in; (void)out_size; (void)ws_size;
    const float* x   = (const float*)d_in[0];
    float*       out = (float*)d_out;

    // Workspace layout (bytes):
    //   [0,256)                      : ordered-uint global max of log_w
    //   [256, 256+16K)               : sq[4096] fp32
    //   [16640, 16640+1MB)           : xn bf16 [4096*128]
    //   [1065216, +64MB)             : masked log_w matrix fp32 [4096*4096] (L2-resident)
    char*     ws       = (char*)d_ws;
    unsigned* gmax_enc = (unsigned*)ws;
    float*    sq       = (float*)(ws + 256);
    bf16_t*   xnb      = (bf16_t*)(ws + 256 + sizeof(float) * NN);
    float*    LW       = (float*)(ws + 1065216);

    k_init<<<1, 64, 0, stream>>>(gmax_enc, out);
    k_normalize<<<NN, DD, 0, stream>>>(x, xnb, sq);
    dim3 g2(NN / 16, NN / (16 * 8)); // 256 tile-cols x 32 row-groups (8 waves/block, 1 tile/wave)
    k_gram<<<g2, 256, 0, stream>>>(xnb, sq, LW, gmax_enc);
    k_sample_loss<<<NN, 256, 0, stream>>>(x, LW, gmax_enc, out);
}
